// SpiralFC1D_65326452572186
// MI455X (gfx1250) — matrix operand order repaired and checked by reading, not compiled, not run on hardware
//
#include <hip/hip_runtime.h>
#include <hip/hip_bf16.h>

// ---------------------------------------------------------------------------
// SpiralFC1D on MI455X (gfx1250):
//   out[b,t,o] = sum_c W[o,c] * X[b,t,row_idx[c,t]] + bias[o]
// row_idx depends only on (c, t%4)  =>  fold gather into 4 effective weight
// matrices Weff[p], then a phase-partitioned f16 WMMA GEMM (f32 accum).
// Wave tile 64x64 (4x4 v_wmma_f32_16x16x32_f16 accumulators) for a 1:1
// load:wmma ratio in the K loop.
// ---------------------------------------------------------------------------

typedef __attribute__((ext_vector_type(16))) _Float16 v16h;
typedef __attribute__((ext_vector_type(8)))  _Float16 v8h;
typedef __attribute__((ext_vector_type(8)))  float    v8f;

#define BATCH 8
#define TLEN  2048
#define CDIM  1024
#define ODIM  1024
#define TPHASE (TLEN / 4)           // 512 rows per (b, phase)
#define MPHASE (BATCH * TPHASE)     // 4096 rows per phase

// ---------------------------------------------------------------------------
// Kernel 1: convert X (f32) -> Xh (f16), same [b,t,c] layout. 8 elts/thread.
// ---------------------------------------------------------------------------
__global__ void spiral_cvt_x(const float* __restrict__ X,
                             _Float16* __restrict__ Xh, long n) {
    long i = ((long)blockIdx.x * blockDim.x + threadIdx.x) * 8;
    if (i >= n) return;
    float4 u = *(const float4*)(X + i);
    float4 v = *(const float4*)(X + i + 4);
    v8h o;
    o[0] = (_Float16)u.x; o[1] = (_Float16)u.y;
    o[2] = (_Float16)u.z; o[3] = (_Float16)u.w;
    o[4] = (_Float16)v.x; o[5] = (_Float16)v.y;
    o[6] = (_Float16)v.z; o[7] = (_Float16)v.w;
    *(v8h*)(Xh + i) = o;
}

// ---------------------------------------------------------------------------
// Kernel 2: build Weff[p][o][j] (f16).
//   d(p) = [0,1,2,1];  c = j + 2 - d(p)
//   Weff = W[o,c] (if 2<=c<C)  +  (j==0)*W[o,0]  +  (j==(p&1))*W[o,1]
// ---------------------------------------------------------------------------
__global__ void spiral_build_weff(const float* __restrict__ W,
                                  _Float16* __restrict__ Weff) {
    int idx = blockIdx.x * blockDim.x + threadIdx.x;
    if (idx >= 4 * ODIM * CDIM) return;
    int j = idx & (CDIM - 1);
    int o = (idx / CDIM) & (ODIM - 1);
    int p = idx / (ODIM * CDIM);
    int d = (p == 2) ? 2 : (p & 1);
    float v = 0.0f;
    int c = j + 2 - d;
    if (c >= 2 && c < CDIM) v += W[o * CDIM + c];
    if (j == 0)             v += W[o * CDIM + 0];
    if (j == ((p & 1) ? 1 : 0)) v += W[o * CDIM + 1];
    Weff[idx] = (_Float16)v;
}

// ---------------------------------------------------------------------------
// Fragment loader: 16 halves from two 16-byte chunks -> v16h (two b128 loads)
// ---------------------------------------------------------------------------
static __device__ __forceinline__ v16h load_frag(const _Float16* p0,
                                                 const _Float16* p1) {
    v8h lo = *(const v8h*)p0;
    v8h hi = *(const v8h*)p1;
    return __builtin_shufflevector(lo, hi, 0, 1, 2, 3, 4, 5, 6, 7,
                                   8, 9, 10, 11, 12, 13, 14, 15);
}

static __device__ __forceinline__ v8f wmma_f16(v16h a, v16h b, v8f c) {
    return __builtin_amdgcn_wmma_f32_16x16x32_f16(
        /*neg_a=*/false, a, /*neg_b=*/false, b,
        /*c_mod=*/(short)0, c, /*reuse_a=*/false, /*reuse_b=*/false);
}

// ---------------------------------------------------------------------------
// Kernel 3: phase-partitioned GEMM.
//   grid = (ODIM/256, MPHASE/128, 4 phases), block = 256 (8 waves, 2M x 4N)
//   wave tile = 64x64 -> 4x4 accumulators; 16 b128 loads : 16 WMMAs per
//   32-wide K step. A rows for phase p are X rows t = p + 4*ti (tiles never
//   cross a batch boundary since 512 % 128 == 0). Weff folds the spiral
//   gather, so A loads are plain strided b128 loads.
// ---------------------------------------------------------------------------
__global__ __launch_bounds__(256) void
spiral_gemm(const _Float16* __restrict__ Xh,
            const _Float16* __restrict__ Weff,
            const float* __restrict__ bias,
            float* __restrict__ out) {
    const int lane = threadIdx.x & 31;
    const int wave = threadIdx.x >> 5;
    const int p    = blockIdx.z;

    const int n0  = blockIdx.x * 256 + (wave & 3) * 64;   // output col base
    const int mi0 = blockIdx.y * 128 + (wave >> 2) * 64;  // phase-row base
    const int bb  = mi0 >> 9;            // batch index (512 rows per batch)
    const int ti0 = mi0 & 511;           // strided time index within batch

    const int lr = lane & 15;            // row/col within 16-group
    const int hi = lane >> 4;            // lane half selector

    // A base pointer (row group 0). Row group g is +16 rows => t += 64
    // => +64*CDIM halves, folded into the instruction offset.
    // 16-bit A layout: lanes 0-15 hold K {0..7,16..23}, lanes 16-31 hold
    // K {8..15,24..31}.
    const _Float16* arow =
        Xh + ((size_t)bb * TLEN + (size_t)(p + 4 * (ti0 + lr))) * CDIM;
    const int aoff = hi * 8;

    // B base pointer (col group 0). Col group g is +16*CDIM halves.
    // B[k,n] = Weff[n,k]: same K layout as A. Lanes 0-15 hold
    // K {0..7,16..23}, lanes 16-31 hold K {8..15,24..31}.
    const _Float16* brow = Weff + ((size_t)p * ODIM + n0 + lr) * CDIM;
    const int boff = hi * 8;

    v8f acc[4][4];
#pragma unroll
    for (int i = 0; i < 4; ++i)
#pragma unroll
        for (int j = 0; j < 4; ++j) acc[i][j] = (v8f){};

#pragma unroll 2
    for (int k = 0; k < CDIM; k += 32) {
        v16h a[4], b[4];
#pragma unroll
        for (int g = 0; g < 4; ++g) {
            const _Float16* pa = arow + (size_t)64 * CDIM * g + k + aoff;
            a[g] = load_frag(pa, pa + 16);
        }
#pragma unroll
        for (int g = 0; g < 4; ++g) {
            const _Float16* pb = brow + (size_t)16 * CDIM * g + k + boff;
            b[g] = load_frag(pb, pb + 16);
        }
#pragma unroll
        for (int i = 0; i < 4; ++i)
#pragma unroll
            for (int j = 0; j < 4; ++j)
                acc[i][j] = wmma_f16(a[i], b[j], acc[i][j]);
    }

    // Epilogue: fuse bias, scatter to out[b,t,o] (t strided by 4 per row).
#pragma unroll
    for (int j = 0; j < 4; ++j) {
        const int ncol = n0 + 16 * j + lr;
        const float bn = bias[ncol];
#pragma unroll
        for (int i = 0; i < 4; ++i) {
#pragma unroll
            for (int v = 0; v < 8; ++v) {
                const int r = v + hi * 8 + 16 * i;     // C/D layout row
                const size_t t0 = (size_t)p + 4 * (size_t)(ti0 + r);
                out[((size_t)bb * TLEN + t0) * ODIM + ncol] = acc[i][j][v] + bn;
            }
        }
    }
}

// ---------------------------------------------------------------------------
extern "C" void kernel_launch(void* const* d_in, const int* in_sizes, int n_in,
                              void* d_out, int out_size, void* d_ws,
                              size_t ws_size, hipStream_t stream) {
    const float* X    = (const float*)d_in[0];   // [8,2048,1024] f32
    const float* W    = (const float*)d_in[1];   // [1024,1024]  f32
    const float* bias = (const float*)d_in[2];   // [1024]       f32
    float* out        = (float*)d_out;           // [8,2048,1024] f32

    // Workspace layout: Xh (32 MiB f16) | Weff (8 MiB f16, 4 phases)
    _Float16* Xh   = (_Float16*)d_ws;
    _Float16* Weff = (_Float16*)((char*)d_ws +
                                 (size_t)BATCH * TLEN * CDIM * sizeof(_Float16));

    const long nX = (long)BATCH * TLEN * CDIM;   // 16,777,216
    spiral_cvt_x<<<(int)(nX / 8 / 256), 256, 0, stream>>>(X, Xh, nX);

    const int nW = 4 * ODIM * CDIM;              // 4,194,304
    spiral_build_weff<<<nW / 256, 256, 0, stream>>>(W, Weff);

    dim3 grid(ODIM / 256, MPHASE / 128, 4);      // (4, 32, 4)
    spiral_gemm<<<grid, 256, 0, stream>>>(Xh, Weff, bias, out);
}
